// LlamaAttention_63574105916115
// MI455X (gfx1250) — matrix operand order repaired and checked by reading, not compiled, not run on hardware
//
#include <hip/hip_runtime.h>

typedef _Float16 f16;
typedef __attribute__((ext_vector_type(8)))  _Float16 v8h;
typedef __attribute__((ext_vector_type(16))) _Float16 v16h;
typedef __attribute__((ext_vector_type(8)))  float    v8f;

constexpr int H = 32, D = 128, HID = 4096, BP = 4, L = 1024, BD = 16, SKV = 512;
constexpr int T = BP * L + BD;
constexpr float SCALE = 0.08838834764831845f;     // 1/sqrt(128)
constexpr float LOG1E4_64 = 0.14391156514261140f; // ln(10000)/64

#define DEVI __device__ __forceinline__

DEVI v16h cat16(v8h lo, v8h hi) {
  return __builtin_shufflevector(lo, hi, 0,1,2,3,4,5,6,7,8,9,10,11,12,13,14,15);
}
DEVI v8f wmma_f16(v16h a, v16h b, v8f c) {
  return __builtin_amdgcn_wmma_f32_16x16x32_f16(false, a, false, b, (short)0, c, false, false);
}
DEVI v8f zero8() { v8f z = {0.f,0.f,0.f,0.f,0.f,0.f,0.f,0.f}; return z; }

// ---- CDNA5 async global->LDS (ASYNCcnt) ----
typedef __attribute__((address_space(3))) const void* as3_cvp;
DEVI unsigned lds_off32(const void* p) {
  return (unsigned)(unsigned long long)(as3_cvp)p;   // addrspacecast -> 32-bit LDS offset
}
DEVI void async_ld_b128(unsigned ldsoff, const void* gaddr) {
  asm volatile("global_load_async_to_lds_b128 %0, %1, off"
               :: "v"(ldsoff), "v"(gaddr) : "memory");
}
DEVI void wait_async0() {
  asm volatile("s_wait_asynccnt 0x0" ::: "memory");
}

// ---------------------------------------------------------------- casts
__global__ void cast_f16_kernel(const float* __restrict__ in, f16* __restrict__ out, size_t n) {
  size_t i = (size_t)blockIdx.x * blockDim.x + threadIdx.x;
  if (i < n) out[i] = (f16)in[i];
}

// ---------------------------------------------------------------- RoPE (in place, q16+k16)
__global__ void rope_qk_kernel(f16* __restrict__ q16, f16* __restrict__ k16) {
  size_t idx = (size_t)blockIdx.x * blockDim.x + threadIdx.x;
  size_t total = (size_t)T * H * 64;
  if (idx >= total) return;
  int i = (int)(idx & 63);
  size_t th = idx >> 6;
  int h = (int)(th % H);
  size_t t = th / H;
  float pos = (t < (size_t)(BP * L)) ? (float)(t % L) : (float)(SKV - 1);
  float inv = __expf(-(float)i * LOG1E4_64);
  float sn, cs;
  __sincosf(pos * inv, &sn, &cs);
  size_t base = t * HID + (size_t)h * D + i;
  {
    float x1 = (float)q16[base], x2 = (float)q16[base + 64];
    q16[base]      = (f16)(x1 * cs - x2 * sn);
    q16[base + 64] = (f16)(x2 * cs + x1 * sn);
  }
  {
    float x1 = (float)k16[base], x2 = (float)k16[base + 64];
    k16[base]      = (f16)(x1 * cs - x2 * sn);
    k16[base + 64] = (f16)(x2 * cs + x1 * sn);
  }
}

// ---------------------------------------------------------------- GEMM  C = A(MxK) * B(NxK)^T
// Double-buffered async global->LDS staging. OUT16: f16 C, else f32 C + bias.
template<bool OUT16, bool BIAS>
__global__ __launch_bounds__(128) void gemm_nt(
    const f16* __restrict__ A, const f16* __restrict__ B,
    void* __restrict__ Cout, const float* __restrict__ bias,
    int M, int N, int K)
{
  __shared__ __align__(16) f16 As[2][128 * 40];
  __shared__ __align__(16) f16 Bs[2][128 * 40];
  const int m0 = blockIdx.y * 128;
  const int n0 = blockIdx.x * 128;
  const int tid  = threadIdx.x;
  const int lane = tid & 31, wave = tid >> 5;
  const int l15 = lane & 15, hs = lane >> 4;
  const int wm = (wave & 1) * 64, wn = (wave >> 1) * 64;
  const int lrow = tid >> 2;   // 0..31
  const int lpart = tid & 3;   // 0..3 (8-half chunks)

  v8f acc[4][4];
#pragma unroll
  for (int i = 0; i < 4; ++i)
#pragma unroll
    for (int j = 0; j < 4; ++j) acc[i][j] = zero8();

  auto issue_stage = [&](int buf, int kb) {
#pragma unroll
    for (int rr = 0; rr < 4; ++rr) {
      int row = lrow + rr * 32;
      int ar = m0 + row; if (ar > M - 1) ar = M - 1;
      async_ld_b128(lds_off32(&As[buf][row * 40 + lpart * 8]),
                    &A[(size_t)ar * K + kb + lpart * 8]);
      async_ld_b128(lds_off32(&Bs[buf][row * 40 + lpart * 8]),
                    &B[(size_t)(n0 + row) * K + kb + lpart * 8]);
    }
  };

  issue_stage(0, 0);
  int buf = 0;
  for (int kb = 0; kb < K; kb += 32) {
    wait_async0();        // own async loads for this stage have landed in LDS
    __syncthreads();      // everyone's landed; everyone done computing other buffer
    if (kb + 32 < K) issue_stage(buf ^ 1, kb + 32);  // overlap with compute below

    v16h af[4], bf[4];
#pragma unroll
    for (int i = 0; i < 4; ++i) {
      const f16* ap = &As[buf][(wm + i * 16 + l15) * 40];
      af[i] = cat16(*(const v8h*)&ap[hs * 8], *(const v8h*)&ap[16 + hs * 8]);
    }
#pragma unroll
    for (int j = 0; j < 4; ++j) {
      const f16* bp = &Bs[buf][(wn + j * 16 + l15) * 40];
      bf[j] = cat16(*(const v8h*)&bp[hs * 8], *(const v8h*)&bp[16 + hs * 8]);
    }
#pragma unroll
    for (int i = 0; i < 4; ++i)
#pragma unroll
      for (int j = 0; j < 4; ++j)
        acc[i][j] = wmma_f16(af[i], bf[j], acc[i][j]);
    buf ^= 1;
  }

#pragma unroll
  for (int j = 0; j < 4; ++j) {
    int col = n0 + wn + j * 16 + l15;
    float bv = 0.f;
    if (BIAS) bv = bias[col];
#pragma unroll
    for (int i = 0; i < 4; ++i) {
      int mbase = m0 + wm + i * 16 + hs * 8;
#pragma unroll
      for (int r = 0; r < 8; ++r) {
        int row = mbase + r;
        if (row < M) {
          if (OUT16) ((f16*)Cout)[(size_t)row * N + col] = (f16)acc[i][j][r];
          else       ((float*)Cout)[(size_t)row * N + col] = acc[i][j][r] + bv;
        }
      }
    }
  }
}

// ---------------------------------------------------------------- prefill flash attention
// grid = (BP*H, L/128), block = 128 (4 waves x 32 query rows)
__global__ __launch_bounds__(128) void prefill_attn(
    const f16* __restrict__ q16, const f16* __restrict__ k16, const f16* __restrict__ v16,
    f16* __restrict__ o16, const int* __restrict__ lens)
{
  __shared__ __align__(16) f16 Vt[128 * 40];  // [d][key], pitch 40
  __shared__ __align__(16) f16 Pb[128 * 40];  // [wave*32+row][key], pitch 40
  const int bh = blockIdx.x;
  const int b = bh >> 5, h = bh & 31;
  const int qb = blockIdx.y;
  const int tid = threadIdx.x;
  const int lane = tid & 31, wave = tid >> 5;
  const int l15 = lane & 15, hs = lane >> 4;
  int len = lens[b]; if (len < 1) len = 1;
  const size_t qrow0 = (size_t)b * L + qb * 128 + wave * 32;

  // preload Q fragments (A-layout)
  v16h qf[2][4];
#pragma unroll
  for (int rt = 0; rt < 2; ++rt)
#pragma unroll
    for (int ks = 0; ks < 4; ++ks) {
      const f16* qp = &q16[(qrow0 + rt * 16 + l15) * HID + h * D + ks * 32];
      qf[rt][ks] = cat16(*(const v8h*)&qp[hs * 8], *(const v8h*)&qp[16 + hs * 8]);
    }

  v8f o[2][8];
  float mrow[2][8], lrow[2][8];
#pragma unroll
  for (int rt = 0; rt < 2; ++rt) {
#pragma unroll
    for (int dt = 0; dt < 8; ++dt) o[rt][dt] = zero8();
#pragma unroll
    for (int r = 0; r < 8; ++r) { mrow[rt][r] = -1e30f; lrow[rt][r] = 0.f; }
  }

  const int kend = (qb + 1) * 128;
  for (int kb = 0; kb < kend; kb += 32) {
    __syncthreads();  // protect Vt from previous iteration's readers
    {   // stage V block transposed into LDS: Vt[d][key]
      int key = tid & 31, dp = tid >> 5;
      const f16* vp = &v16[((size_t)b * L + kb + key) * HID + h * D + dp * 32];
#pragma unroll
      for (int c = 0; c < 4; ++c) {
        v8h vv = *(const v8h*)&vp[c * 8];
#pragma unroll
        for (int e = 0; e < 8; ++e) Vt[(dp * 32 + c * 8 + e) * 40 + key] = vv[e];
      }
    }
    __syncthreads();

    // scores S = Q K^T for 2 row-tiles x 2 key-tiles
    v8f s[2][2];
#pragma unroll
    for (int kt = 0; kt < 2; ++kt) {
      v16h kf[4];
#pragma unroll
      for (int ks = 0; ks < 4; ++ks) {
        const f16* kp = &k16[((size_t)b * L + kb + kt * 16 + l15) * HID + h * D + ks * 32];
        kf[ks] = cat16(*(const v8h*)&kp[hs * 8], *(const v8h*)&kp[16 + hs * 8]);
      }
#pragma unroll
      for (int rt = 0; rt < 2; ++rt) {
        v8f t = zero8();
#pragma unroll
        for (int ks = 0; ks < 4; ++ks) t = wmma_f16(qf[rt][ks], kf[ks], t);
        s[rt][kt] = t;
      }
    }

    // online softmax update per row-tile
#pragma unroll
    for (int rt = 0; rt < 2; ++rt) {
      float t0[8], t1[8];
#pragma unroll
      for (int r = 0; r < 8; ++r) {
        int qr = qb * 128 + wave * 32 + rt * 16 + r + hs * 8;
        int k0 = kb + l15, k1 = kb + 16 + l15;
        float a0 = s[rt][0][r] * SCALE, a1 = s[rt][1][r] * SCALE;
        t0[r] = (k0 <= qr && k0 < len) ? a0 : -1e30f;
        t1[r] = (k1 <= qr && k1 < len) ? a1 : -1e30f;
      }
#pragma unroll
      for (int r = 0; r < 8; ++r) {
        float mx = fmaxf(t0[r], t1[r]);
        for (int off = 8; off; off >>= 1) mx = fmaxf(mx, __shfl_xor(mx, off));
        float mo = mrow[rt][r];
        float mn = fmaxf(mo, mx);
        float alpha = __expf(mo - mn);
        mrow[rt][r] = mn;
        float p0 = __expf(t0[r] - mn), p1 = __expf(t1[r] - mn);
        float rs = p0 + p1;
        for (int off = 8; off; off >>= 1) rs += __shfl_xor(rs, off);
        lrow[rt][r] = lrow[rt][r] * alpha + rs;
#pragma unroll
        for (int dt = 0; dt < 8; ++dt) o[rt][dt][r] *= alpha;
        int prow = wave * 32 + rt * 16 + r + hs * 8;
        Pb[prow * 40 + l15]      = (f16)p0;
        Pb[prow * 40 + 16 + l15] = (f16)p1;
      }
    }

    // O += P * V   (P via LDS C->A layout, V^T from Vt)
#pragma unroll
    for (int rt = 0; rt < 2; ++rt) {
      const f16* pp = &Pb[(wave * 32 + rt * 16 + l15) * 40];
      v16h pf = cat16(*(const v8h*)&pp[hs * 8], *(const v8h*)&pp[16 + hs * 8]);
#pragma unroll
      for (int dt = 0; dt < 8; ++dt) {
        const f16* vp = &Vt[(dt * 16 + l15) * 40];
        v16h bfr = cat16(*(const v8h*)&vp[hs * 8], *(const v8h*)&vp[16 + hs * 8]);
        o[rt][dt] = wmma_f16(pf, bfr, o[rt][dt]);
      }
    }
  }

  // normalize, zero padded query rows, store f16
#pragma unroll
  for (int rt = 0; rt < 2; ++rt)
#pragma unroll
    for (int dt = 0; dt < 8; ++dt)
#pragma unroll
      for (int r = 0; r < 8; ++r) {
        int qr = qb * 128 + wave * 32 + rt * 16 + r + hs * 8;
        float val = (qr < len) ? (o[rt][dt][r] / lrow[rt][r]) : 0.f;
        o16[((size_t)b * L + qr) * HID + h * D + dt * 16 + l15] = (f16)val;
      }
}

// ---------------------------------------------------------------- decode attention
// grid = BD*H, block = 256
__global__ __launch_bounds__(256) void decode_attn(
    const f16* __restrict__ q16, const f16* __restrict__ k16, const f16* __restrict__ v16,
    const float* __restrict__ kvk, const float* __restrict__ kvv,
    f16* __restrict__ o16)
{
  __shared__ float qs[D];
  __shared__ float invf[64];
  __shared__ float sc[SKV];
  __shared__ float red[8];
  __shared__ float stat[2];
  __shared__ float ob[D];
  const int bh = blockIdx.x;
  const int b = bh >> 5, h = bh & 31;
  const int tid = threadIdx.x;
  const int lane = tid & 31, wv = tid >> 5;
  const size_t drow = (size_t)(BP * L + b);

  if (tid < D) qs[tid] = (float)q16[drow * HID + h * D + tid];
  if (tid >= 128 && tid < 192) invf[tid - 128] = __expf(-(float)(tid - 128) * LOG1E4_64);
  __syncthreads();

  for (int s = tid; s < SKV; s += 256) {
    float dot = 0.f;
    if (s == SKV - 1) {
      const f16* kp = &k16[drow * HID + h * D];  // already RoPE'd at pos 511
#pragma unroll 16
      for (int d = 0; d < D; ++d) dot += qs[d] * (float)kp[d];
    } else {
      const float* kp = &kvk[(((size_t)b * (SKV - 1) + s) * H + h) * D];
      float pos = (float)s;
      for (int i = 0; i < 64; ++i) {
        float sn, cs; __sincosf(pos * invf[i], &sn, &cs);
        float x1 = kp[i], x2 = kp[i + 64];
        dot += qs[i] * (x1 * cs - x2 * sn) + qs[i + 64] * (x2 * cs + x1 * sn);
      }
    }
    sc[s] = dot * SCALE;
  }
  __syncthreads();

  float m = -1e30f;
  for (int s = tid; s < SKV; s += 256) m = fmaxf(m, sc[s]);
  for (int off = 16; off; off >>= 1) m = fmaxf(m, __shfl_xor(m, off));
  if (lane == 0) red[wv] = m;
  __syncthreads();
  if (tid == 0) {
    float M = red[0];
    for (int w = 1; w < 8; ++w) M = fmaxf(M, red[w]);
    stat[0] = M;
  }
  __syncthreads();
  const float M = stat[0];
  float ls = 0.f;
  for (int s = tid; s < SKV; s += 256) { float pe = __expf(sc[s] - M); sc[s] = pe; ls += pe; }
  for (int off = 16; off; off >>= 1) ls += __shfl_xor(ls, off);
  if (lane == 0) red[wv] = ls;
  __syncthreads();
  if (tid == 0) {
    float Lt = 0.f;
    for (int w = 0; w < 8; ++w) Lt += red[w];
    stat[1] = Lt;
  }
  __syncthreads();
  const float Ls = stat[1];

  const int d = tid & (D - 1);
  const int halfsel = tid >> 7;
  float acc = 0.f;
  for (int s = halfsel * 256; s < halfsel * 256 + 256; ++s) {
    float vv;
    if (s == SKV - 1) vv = (float)v16[drow * HID + h * D + d];
    else              vv = kvv[(((size_t)b * (SKV - 1) + s) * H + h) * D + d];
    acc += sc[s] * vv;
  }
  if (halfsel == 0) ob[d] = acc;
  __syncthreads();
  if (halfsel == 1) o16[drow * HID + h * D + d] = (f16)((ob[d] + acc) / Ls);
}

// ---------------------------------------------------------------- host launcher
extern "C" void kernel_launch(void* const* d_in, const int* in_sizes, int n_in,
                              void* d_out, int out_size, void* d_ws, size_t ws_size,
                              hipStream_t stream) {
  (void)in_sizes; (void)n_in; (void)out_size; (void)ws_size;
  const float* hidden = (const float*)d_in[0];
  const float* Wq = (const float*)d_in[1];
  const float* Wk = (const float*)d_in[2];
  const float* Wv = (const float*)d_in[3];
  const float* Wo = (const float*)d_in[4];
  const float* bias = (const float*)d_in[5];
  const float* kvk = (const float*)d_in[6];
  const float* kvv = (const float*)d_in[7];
  const int*   lens = (const int*)d_in[8];

  const size_t szTH = (size_t)T * HID;
  const size_t szW  = (size_t)HID * HID;
  f16* p = (f16*)d_ws;
  f16* h16  = p; p += szTH;
  f16* wq16 = p; p += szW;
  f16* wk16 = p; p += szW;
  f16* wv16 = p; p += szW;
  f16* wo16 = p; p += szW;
  f16* q16  = p; p += szTH;
  f16* k16  = p; p += szTH;
  f16* v16  = p; p += szTH;
  f16* o16  = p; p += szTH;

  auto cdiv = [](size_t a, size_t b) { return (unsigned)((a + b - 1) / b); };

  cast_f16_kernel<<<cdiv(szTH, 512), 512, 0, stream>>>(hidden, h16, szTH);
  cast_f16_kernel<<<cdiv(szW, 512), 512, 0, stream>>>(Wq, wq16, szW);
  cast_f16_kernel<<<cdiv(szW, 512), 512, 0, stream>>>(Wk, wk16, szW);
  cast_f16_kernel<<<cdiv(szW, 512), 512, 0, stream>>>(Wv, wv16, szW);
  cast_f16_kernel<<<cdiv(szW, 512), 512, 0, stream>>>(Wo, wo16, szW);

  dim3 gg(HID / 128, (T + 127) / 128);
  gemm_nt<true, false><<<gg, 128, 0, stream>>>(h16, wq16, (void*)q16, nullptr, T, HID, HID);
  gemm_nt<true, false><<<gg, 128, 0, stream>>>(h16, wk16, (void*)k16, nullptr, T, HID, HID);
  gemm_nt<true, false><<<gg, 128, 0, stream>>>(h16, wv16, (void*)v16, nullptr, T, HID, HID);

  size_t nrope = (size_t)T * H * 64;
  rope_qk_kernel<<<cdiv(nrope, 256), 256, 0, stream>>>(q16, k16);

  prefill_attn<<<dim3(BP * H, L / 128), 128, 0, stream>>>(q16, k16, v16, o16, lens);
  decode_attn<<<dim3(BD * H), 256, 0, stream>>>(q16, k16, v16, kvk, kvv, o16);

  gemm_nt<false, true><<<gg, 128, 0, stream>>>(o16, wo16, d_out, bias, T, HID, HID);
}
